// GraphSpectralConv_6107443494942
// MI455X (gfx1250) — matrix operand order repaired and checked by reading, not compiled, not run on hardware
//
#include <hip/hip_runtime.h>

// ---------------------------------------------------------------------------
// GraphSpectralConv for MI455X (gfx1250, wave32, WMMA)
//   phase1: partial[b,kc,mode,c] = sum_{n in chunk kc} U[b,n,mode] * h[b,n,c]
//   phase2: out_hat[b,mode,o]    = sum_c (sum_kc partial) * W[mode,c,o]   (f16 out)
//   phase3: out[b,n,o]           = sum_mode U[b,n,mode] * out_hat[b,mode,o]
// All matmuls via v_wmma_f32_16x16x32_f16 (f32 accumulate).
// ---------------------------------------------------------------------------

typedef __attribute__((ext_vector_type(16))) _Float16     v16h;
typedef __attribute__((ext_vector_type(8)))  float        v8f;
typedef __attribute__((ext_vector_type(4)))  unsigned int u32x4;

union V16H { v16h v; u32x4 q[2]; };

#define BATCH   128
#define NNODES  4096
#define MODES   64
#define CH      128
#define KCHUNKS 4
#define KCSZ    (NNODES / KCHUNKS)   // 1024

// LDS row strides (in halves). 40 halves = 80 B: rows start on 16B boundaries
// and row-start banks are spaced exactly 4 apart -> conflict-free b128 loads.
#define S40 40
// 72 halves = 144 B for K=64 rows in phase 3 (also 16B aligned, spacing 4).
#define S72 72

__device__ __forceinline__ v8f wmma_f16(const V16H& a, const V16H& b, v8f c) {
    return __builtin_amdgcn_wmma_f32_16x16x32_f16(
        /*neg_a=*/false, a.v, /*neg_b=*/false, b.v,
        /*c_mod=*/(short)0, c, /*reuse_a=*/false, /*reuse_b=*/false);
}

// -------------------------------------------------------------- phase 1 ----
// grid: (KCHUNKS, BATCH), block: 256 (8 waves)
// Each block: batch b, nodes [kc*1024, kc*1024+1024), output 64x128 f32 tile.
// Wave w owns modes [16*(w>>1), +16) x channels [64*(w&1), +64)  (4 tiles).
__global__ __launch_bounds__(256) void
phase1_uth(const float* __restrict__ h, const float* __restrict__ U,
           float* __restrict__ partial) {
    const int b    = blockIdx.y;
    const int kc   = blockIdx.x;
    const int t    = threadIdx.x;
    const int lane = t & 31;
    const int wave = t >> 5;

    __shared__ __align__(16) _Float16 Ut[MODES * S40]; // [mode][node]  (A, transposed)
    __shared__ __align__(16) _Float16 Ht[CH    * S40]; // [ch]  [node]  (B, transposed)

    const int mbase = (wave >> 1) * 16;
    const int cbase = (wave & 1) * 64;

    v8f acc[4] = {};

    const size_t ubase = ((size_t)b * NNODES + (size_t)kc * KCSZ) * MODES;
    const size_t hbase = ((size_t)b * NNODES + (size_t)kc * KCSZ) * CH;

    for (int k0 = 0; k0 < KCSZ; k0 += 32) {
        // stage U chunk: 32 nodes x 64 modes (mode fastest in global -> coalesced)
#pragma unroll
        for (int i = 0; i < 8; ++i) {
            int idx  = t + 256 * i;          // 2048 elements
            int node = idx >> 6, mode = idx & 63;
            Ut[mode * S40 + node] =
                (_Float16)U[ubase + (size_t)(k0 + node) * MODES + mode];
        }
        // stage h chunk: 32 nodes x 128 ch
#pragma unroll
        for (int i = 0; i < 16; ++i) {
            int idx  = t + 256 * i;          // 4096 elements
            int node = idx >> 7, ch = idx & 127;
            Ht[ch * S40 + node] =
                (_Float16)h[hbase + (size_t)(k0 + node) * CH + ch];
        }
        __syncthreads();

        // A fragment: row m = mode, K = node (ISA 16-bit A layout)
        const int m  = mbase + (lane & 15);
        const int hi = lane >> 4;
        V16H a;
        a.q[0] = *(const u32x4*)&Ut[m * S40 + 8 * hi];        // K = 8*hi + 0..7
        a.q[1] = *(const u32x4*)&Ut[m * S40 + 16 + 8 * hi];   // K = 16 + 8*hi + 0..7

#pragma unroll
        for (int ts = 0; ts < 4; ++ts) {
            const int c = cbase + ts * 16 + (lane & 15);
            V16H bb;                                          // K = 8*hi + 0..7, 16 + 8*hi + 0..7
            bb.q[0] = *(const u32x4*)&Ht[c * S40 + 8 * hi];
            bb.q[1] = *(const u32x4*)&Ht[c * S40 + 16 + 8 * hi];
            acc[ts] = wmma_f16(a, bb, acc[ts]);
        }
        __syncthreads();
    }

    // store f32 partials: C layout -> lane n = col, VGPR j -> row j + 8*hi
    float* pout = partial + (((size_t)b * KCHUNKS + kc) * MODES) * CH;
    const int hi  = lane >> 4;
    const int n15 = lane & 15;
#pragma unroll
    for (int ts = 0; ts < 4; ++ts) {
        const int c = cbase + ts * 16 + n15;
#pragma unroll
        for (int j = 0; j < 8; ++j) {
            const int mrow = mbase + j + 8 * hi;
            pout[(size_t)mrow * CH + c] = acc[ts][j];
        }
    }
}

// -------------------------------------------------------------- phase 2 ----
// grid: (MODES), block: 256. Per mode: [B=128 x C=128(K)] @ W[mode] -> 128x128.
// Wave w owns batch rows [16w, +16) x all 8 output strips.
__global__ __launch_bounds__(256) void
phase2_mix(const float* __restrict__ partial, const float* __restrict__ W,
           _Float16* __restrict__ outhat) {
    const int mode = blockIdx.x;
    const int t    = threadIdx.x;
    const int lane = t & 31;
    const int wave = t >> 5;

    __shared__ __align__(16) _Float16 At[BATCH * S40]; // [b][c]  (A, K=c)
    __shared__ __align__(16) _Float16 Wt[CH    * S40]; // [o][c]  (B, transposed)

    v8f acc[8] = {};
    const int bbase = wave * 16;

    for (int k0 = 0; k0 < CH; k0 += 32) {
        // stage A: reduce the 4 K-chunk partials on the fly (128 b x 32 c)
#pragma unroll
        for (int i = 0; i < 16; ++i) {
            int idx = t + 256 * i;           // 4096 elements
            int bb  = idx >> 5, c = idx & 31;
            float s = 0.f;
#pragma unroll
            for (int kc = 0; kc < KCHUNKS; ++kc)
                s += partial[(((size_t)bb * KCHUNKS + kc) * MODES + mode) * CH
                             + (size_t)(k0 + c)];
            At[bb * S40 + c] = (_Float16)s;
        }
        // stage W[mode]: 32 c x 128 o (o fastest -> coalesced), transpose
#pragma unroll
        for (int i = 0; i < 16; ++i) {
            int idx = t + 256 * i;
            int c   = idx >> 7, o = idx & 127;
            Wt[o * S40 + c] =
                (_Float16)W[((size_t)mode * CH + (size_t)(k0 + c)) * CH + o];
        }
        __syncthreads();

        const int r  = bbase + (lane & 15);
        const int hi = lane >> 4;
        V16H a;
        a.q[0] = *(const u32x4*)&At[r * S40 + 8 * hi];
        a.q[1] = *(const u32x4*)&At[r * S40 + 16 + 8 * hi];

#pragma unroll
        for (int ts = 0; ts < 8; ++ts) {
            const int o = ts * 16 + (lane & 15);
            V16H bb;
            bb.q[0] = *(const u32x4*)&Wt[o * S40 + 8 * hi];
            bb.q[1] = *(const u32x4*)&Wt[o * S40 + 16 + 8 * hi];
            acc[ts] = wmma_f16(a, bb, acc[ts]);
        }
        __syncthreads();
    }

    // store out_hat as f16 (consumed as B-operand by phase 3)
    const int hi  = lane >> 4;
    const int n15 = lane & 15;
#pragma unroll
    for (int ts = 0; ts < 8; ++ts) {
        const int o = ts * 16 + n15;
#pragma unroll
        for (int j = 0; j < 8; ++j) {
            const int brow = bbase + j + 8 * hi;
            outhat[((size_t)brow * MODES + mode) * CH + o] = (_Float16)acc[ts][j];
        }
    }
}

// -------------------------------------------------------------- phase 3 ----
// grid: (32 node tiles, BATCH), block 256. Tile: 128 nodes x 128 out, K=64.
// Wave w owns node rows [16w, +16) x all 8 output strips.
__global__ __launch_bounds__(256) void
phase3_back(const float* __restrict__ U, const _Float16* __restrict__ outhat,
            float* __restrict__ out) {
    const int b        = blockIdx.y;
    const int nodeBase = blockIdx.x * 128;
    const int t        = threadIdx.x;
    const int lane     = t & 31;
    const int wave     = t >> 5;

    __shared__ __align__(16) _Float16 Ut3[128 * S72]; // [node][mode] (A, K=mode)
    __shared__ __align__(16) _Float16 Bt3[CH  * S72]; // [o][mode]    (B, transposed)

    // stage U tile: 128 nodes x 64 modes
#pragma unroll
    for (int i = 0; i < 32; ++i) {
        int idx  = t + 256 * i;              // 8192 elements
        int node = idx >> 6, mode = idx & 63;
        Ut3[node * S72 + mode] =
            (_Float16)U[((size_t)b * NNODES + nodeBase + node) * MODES + mode];
    }
    // stage out_hat[b]: 64 modes x 128 o  ->  transpose to [o][mode]
#pragma unroll
    for (int i = 0; i < 32; ++i) {
        int idx = t + 256 * i;
        int mu  = idx >> 7, o = idx & 127;
        Bt3[o * S72 + mu] = outhat[((size_t)b * MODES + mu) * CH + o];
    }
    __syncthreads();

    v8f acc[8] = {};
    const int m  = wave * 16 + (lane & 15);
    const int hi = lane >> 4;
    const int n15 = lane & 15;

#pragma unroll
    for (int kk = 0; kk < 2; ++kk) {         // K = 64 -> two x32 steps
        V16H a;
        a.q[0] = *(const u32x4*)&Ut3[m * S72 + kk * 32 + 8 * hi];
        a.q[1] = *(const u32x4*)&Ut3[m * S72 + kk * 32 + 16 + 8 * hi];
#pragma unroll
        for (int ts = 0; ts < 8; ++ts) {
            const int o = ts * 16 + n15;
            V16H bb;
            bb.q[0] = *(const u32x4*)&Bt3[o * S72 + kk * 32 + 8 * hi];
            bb.q[1] = *(const u32x4*)&Bt3[o * S72 + kk * 32 + 16 + 8 * hi];
            acc[ts] = wmma_f16(a, bb, acc[ts]);
        }
    }

#pragma unroll
    for (int ts = 0; ts < 8; ++ts) {
        const int o = ts * 16 + n15;
#pragma unroll
        for (int j = 0; j < 8; ++j) {
            const int node = nodeBase + wave * 16 + j + 8 * hi;
            out[((size_t)b * NNODES + node) * CH + o] = acc[ts][j];
        }
    }
}

// ---------------------------------------------------------------------------
extern "C" void kernel_launch(void* const* d_in, const int* in_sizes, int n_in,
                              void* d_out, int out_size, void* d_ws, size_t ws_size,
                              hipStream_t stream) {
    (void)in_sizes; (void)n_in; (void)out_size; (void)ws_size;
    const float* h = (const float*)d_in[0];   // [B*N, C] f32
    const float* U = (const float*)d_in[1];   // [B, N, m] f32
    const float* W = (const float*)d_in[2];   // [m, C, C] f32
    // d_in[3] = ptr (equal-size segments; layout is dense [B, N]) -> unused
    float* out = (float*)d_out;

    const size_t partial_bytes = (size_t)BATCH * KCHUNKS * MODES * CH * sizeof(float); // 16 MB
    float*    partial = (float*)d_ws;
    _Float16* outhat  = (_Float16*)((char*)d_ws + partial_bytes);                      // +2 MB

    phase1_uth <<<dim3(KCHUNKS, BATCH), 256, 0, stream>>>(h, U, partial);
    phase2_mix <<<dim3(MODES),          256, 0, stream>>>(partial, W, outhat);
    phase3_back<<<dim3(NNODES / 128, BATCH), 256, 0, stream>>>(U, outhat, out);
}
